// MultiHeadedSelfAttention_30726196036187
// MI455X (gfx1250) — matrix operand order repaired and checked by reading, not compiled, not run on hardware
//
#include <hip/hip_runtime.h>
#include <stdint.h>

#define D_  1024
#define H_  16
#define HD_ 64
#define B_  2
#define L_  2048

typedef __attribute__((ext_vector_type(16))) __bf16   v16bf;
typedef __attribute__((ext_vector_type(8)))  float    v8f;
typedef __attribute__((ext_vector_type(4)))  uint32_t v4u;
typedef __attribute__((ext_vector_type(8)))  int      v8i;
typedef __attribute__((ext_vector_type(4)))  int      v4i;

// Staged LDS tiles get a 1-dword pad every 16 dwords (TDM pad feature):
// row stride = 32 bf16 + 2 = 34 elements (17 dwords, coprime with 64 banks).
#define LDS_STRIDE 34

// ---------- helpers ----------

// float -> bf16, round-to-nearest-even via bit trick
__device__ inline __bf16 f2bf(float f) {
    union { float f; uint32_t u; } x; x.f = f;
    uint32_t r = (x.u + 0x7FFFu + ((x.u >> 16) & 1u)) >> 16;
    union { unsigned short s; __bf16 b; } y; y.s = (unsigned short)r;
    return y.b;
}

__device__ inline v8f wmma_bf16(v16bf a, v16bf b, v8f c) {
    return __builtin_amdgcn_wmma_f32_16x16x32_bf16(
        false, a, false, b, (short)0, c, false, false);
}

// A-fragment 16x32 bf16 (M x K), row-major source `tile` (stride lda elements).
__device__ inline v16bf load_a_frag(const __bf16* tile, int lda) {
    int lane = threadIdx.x & 31;
    int m  = lane & 15;
    int kb = (lane & 16) ? 16 : 0;
    const __bf16* p = tile + (size_t)m * lda + kb;
    union { v16bf v; uint32_t u[8]; } r;
#pragma unroll
    for (int i = 0; i < 8; ++i) r.u[i] = *(const uint32_t*)(p + 2 * i);
    return r.v;
}

// B-fragment 32x16 bf16 (K x N) from an N-major source: tileT[n][k], stride ldb.
__device__ inline v16bf load_b_fragT(const __bf16* tileT, int ldb) {
    int lane = threadIdx.x & 31;
    int n  = lane & 15;
    int kb = (lane & 16) ? 16 : 0;
    const __bf16* p = tileT + (size_t)n * ldb + kb;
    union { v16bf v; uint32_t u[8]; } r;
#pragma unroll
    for (int i = 0; i < 8; ++i) r.u[i] = *(const uint32_t*)(p + 2 * i);
    return r.v;
}

// ---------- Tensor Data Mover: 2D tile global -> LDS ----------
// D# per CDNA5 ISA. data_size=2B, pad every 16 dwords by
// 1 dword (-> 34-element LDS row stride), count=1, type=2 ("image").
__device__ inline void tdm_load_2d(uint32_t lds_addr, const void* gaddr,
                                   uint32_t tensor_d0, uint32_t tensor_d1,
                                   uint32_t stride0_elems,
                                   uint32_t tile_d0, uint32_t tile_d1) {
    uint64_t ga = (uint64_t)(uintptr_t)gaddr;
    v4u g0;
    g0[0] = 1u;                                      // count=1
    g0[1] = lds_addr;                                // lds_addr [63:32]
    g0[2] = (uint32_t)ga;                            // global_addr lo
    g0[3] = (uint32_t)((ga >> 32) & 0x01FFFFFFu)     // global_addr hi
          | 0x80000000u;                             // type=2 at bits [127:126]
    v8i g1;
    g1[0] = (int)(0x00010000u                        // data_size=1 (2 bytes)
          | (1u << 20)                               // pad_enable
          | (3u << 22));                             // pad_interval=3 (16 dwords); pad_amount=0 (1 dword)
    g1[1] = (int)((tensor_d0 & 0xFFFFu) << 16);                                  // tensor_dim0[15:0]
    g1[2] = (int)(((tensor_d0 >> 16) & 0xFFFFu) | ((tensor_d1 & 0xFFFFu) << 16)); // dim0 hi | dim1 lo
    g1[3] = (int)(((tensor_d1 >> 16) & 0xFFFFu) | ((tile_d0 & 0xFFFFu) << 16));   // dim1 hi | tile_dim0
    g1[4] = (int)(tile_d1 & 0xFFFFu);                // tile_dim1 (tile_dim2=0)
    g1[5] = (int)stride0_elems;                      // tensor_dim0_stride[31:0]
    g1[6] = 0;
    g1[7] = 0;
    v4i z4 = {0, 0, 0, 0};
#if __clang_major__ >= 23
    v8i z8 = {0, 0, 0, 0, 0, 0, 0, 0};
    __builtin_amdgcn_tensor_load_to_lds(g0, g1, z4, z4, z8, 0);
#else
    __builtin_amdgcn_tensor_load_to_lds(g0, g1, z4, z4, 0);
#endif
}

__device__ inline uint32_t lds_off(const void* p) {
    return (uint32_t)(uintptr_t)p;   // low 32 bits of generic addr = LDS offset
}

// ---------- kernel 0: fp32 -> bf16 conversion + weight transposes ----------

__global__ __launch_bounds__(256)
void mhsa_convert(const float* __restrict__ X,  const float* __restrict__ Wq,
                  const float* __restrict__ Wk, const float* __restrict__ Wv,
                  const float* __restrict__ Wo,
                  __bf16* __restrict__ Xbf,  __bf16* __restrict__ Wqt,
                  __bf16* __restrict__ Wkt,  __bf16* __restrict__ Wvt,
                  __bf16* __restrict__ WoT) {
    size_t tid = (size_t)blockIdx.x * blockDim.x + threadIdx.x;
    size_t gs  = (size_t)gridDim.x * blockDim.x;
    for (size_t i = tid; i < (size_t)B_ * L_ * D_; i += gs)
        Xbf[i] = f2bf(X[i]);
    for (size_t i = tid; i < (size_t)H_ * D_ * HD_; i += gs) {
        int h = (int)(i / (D_ * HD_));
        int rem = (int)(i % (D_ * HD_));
        int d = rem / HD_, e = rem % HD_;
        size_t dst = (size_t)h * HD_ * D_ + (size_t)e * D_ + d;
        Wqt[dst] = f2bf(Wq[i]);
        Wkt[dst] = f2bf(Wk[i]);
        Wvt[dst] = f2bf(Wv[i]);
    }
    for (size_t i = tid; i < (size_t)D_ * D_; i += gs) {
        int k = (int)(i / D_), n = (int)(i % D_);
        WoT[(size_t)n * D_ + k] = f2bf(Wo[i]);
    }
}

// ---------- kernel 1: fused QKV projection, TDM-staged, double-buffered ----------
// grid = (B*H, L/128), block = 256 (8 waves). Block stages X tile (128x32) and
// Wq/Wk/Wv tiles (64x32) into LDS via TDM; each wave owns a 16-row strip.

__global__ __launch_bounds__(256)
void mhsa_qkv(const __bf16* __restrict__ Xbf,
              const __bf16* __restrict__ Wqt, const __bf16* __restrict__ Wkt,
              const __bf16* __restrict__ Wvt,
              __bf16* __restrict__ Q, __bf16* __restrict__ K, __bf16* __restrict__ Vt) {
    __shared__ __bf16 ldsX[2][128 * LDS_STRIDE];       // 2 x 8704 B
    __shared__ __bf16 ldsW[2][3][64 * LDS_STRIDE];     // 2 x 3 x 4352 B

    int bh   = blockIdx.x;
    int b    = bh / H_;
    int h    = bh % H_;
    int wave = threadIdx.x >> 5;
    int lane = threadIdx.x & 31;
    int row0 = blockIdx.y * 128;

    const __bf16* Xbase = Xbf + ((size_t)(b * L_ + row0)) * D_;
    const __bf16* W0 = Wqt + (size_t)h * HD_ * D_;
    const __bf16* W1 = Wkt + (size_t)h * HD_ * D_;
    const __bf16* W2 = Wvt + (size_t)h * HD_ * D_;

    v8f aq[4] = {}, ak[4] = {}, av[4] = {};

    if (wave == 0) {   // prologue: fill buffer 0
        tdm_load_2d(lds_off(&ldsX[0][0]),    Xbase, D_, 1u << 20, D_, 32, 128);
        tdm_load_2d(lds_off(&ldsW[0][0][0]), W0,    D_, 1u << 20, D_, 32, 64);
        tdm_load_2d(lds_off(&ldsW[0][1][0]), W1,    D_, 1u << 20, D_, 32, 64);
        tdm_load_2d(lds_off(&ldsW[0][2][0]), W2,    D_, 1u << 20, D_, 32, 64);
    }

    for (int ks = 0; ks < D_ / 32; ++ks) {
        int buf = ks & 1;
        if (wave == 0) {
            if (ks + 1 < D_ / 32) {
                int k0n = (ks + 1) * 32;
                tdm_load_2d(lds_off(&ldsX[buf ^ 1][0]),    Xbase + k0n, D_, 1u << 20, D_, 32, 128);
                tdm_load_2d(lds_off(&ldsW[buf ^ 1][0][0]), W0 + k0n,    D_, 1u << 20, D_, 32, 64);
                tdm_load_2d(lds_off(&ldsW[buf ^ 1][1][0]), W1 + k0n,    D_, 1u << 20, D_, 32, 64);
                tdm_load_2d(lds_off(&ldsW[buf ^ 1][2][0]), W2 + k0n,    D_, 1u << 20, D_, 32, 64);
                __builtin_amdgcn_s_wait_tensorcnt(4);   // current buf's 4 ops done
            } else {
                __builtin_amdgcn_s_wait_tensorcnt(0);
            }
        }
        __syncthreads();

        v16bf a = load_a_frag(&ldsX[buf][wave * 16 * LDS_STRIDE], LDS_STRIDE);
#pragma unroll
        for (int nt = 0; nt < 4; ++nt) {
            const __bf16* wb = &ldsW[buf][0][nt * 16 * LDS_STRIDE];
            aq[nt] = wmma_bf16(a, load_b_fragT(&ldsW[buf][0][nt * 16 * LDS_STRIDE], LDS_STRIDE), aq[nt]);
            ak[nt] = wmma_bf16(a, load_b_fragT(&ldsW[buf][1][nt * 16 * LDS_STRIDE], LDS_STRIDE), ak[nt]);
            av[nt] = wmma_bf16(a, load_b_fragT(&ldsW[buf][2][nt * 16 * LDS_STRIDE], LDS_STRIDE), av[nt]);
            (void)wb;
        }
        __syncthreads();   // all waves done with buf before it is refilled
    }

    int n  = lane & 15;
    int mh = (lane & 16) ? 8 : 0;
    int rowW = row0 + wave * 16;
    size_t qb = (size_t)bh * L_ * HD_ + (size_t)rowW * HD_;
    size_t vb = (size_t)bh * HD_ * L_;
#pragma unroll
    for (int nt = 0; nt < 4; ++nt) {
#pragma unroll
        for (int r = 0; r < 8; ++r) {
            int m = mh + r, e = nt * 16 + n;
            Q[qb + (size_t)m * HD_ + e] = f2bf(aq[nt][r] * 0.125f);  // 1/sqrt(64)
            K[qb + (size_t)m * HD_ + e] = f2bf(ak[nt][r]);
            Vt[vb + (size_t)e * L_ + (rowW + m)] = f2bf(av[nt][r]);
        }
    }
}

// ---------- kernel 2: causal flash attention ----------

__global__ __launch_bounds__(256)
void mhsa_attn(const __bf16* __restrict__ Q, const __bf16* __restrict__ K,
               const __bf16* __restrict__ Vt, __bf16* __restrict__ AO) {
    __shared__ __bf16 plds[8][16 * 34];   // per-wave 16x32 P tile, stride 34

    int bh   = blockIdx.x;
    int b    = bh / H_;
    int h    = bh % H_;
    int wave = threadIdx.x >> 5;
    int lane = threadIdx.x & 31;
    int q0   = blockIdx.y * 128 + wave * 16;

    const __bf16* Qb = Q  + (size_t)bh * L_ * HD_;
    const __bf16* Kb = K  + (size_t)bh * L_ * HD_;
    const __bf16* Vb = Vt + (size_t)bh * HD_ * L_;   // [e][l]
    __bf16* myp = &plds[wave][0];

    v16bf qa0 = load_a_frag(Qb + (size_t)q0 * HD_ + 0,  HD_);
    v16bf qa1 = load_a_frag(Qb + (size_t)q0 * HD_ + 32, HD_);

    v8f o[4] = {};
    float mrow[8], lrow[8];
#pragma unroll
    for (int r = 0; r < 8; ++r) { mrow[r] = -1e30f; lrow[r] = 0.0f; }

    int n  = lane & 15;
    int mh = (lane & 16) ? 8 : 0;

    for (int j = 0; j <= q0; j += 32) {           // keys [j, j+32)
        if (j + 32 <= q0) {
            __builtin_prefetch(Kb + (size_t)(j + 32) * HD_, 0, 1);
            __builtin_prefetch(Vb + (j + 32), 0, 1);
        }
        v8f s0 = {}, s1 = {};
        {
            v16bf b0 = load_b_fragT(Kb + (size_t)j * HD_ + 0,  HD_);
            v16bf b1 = load_b_fragT(Kb + (size_t)j * HD_ + 32, HD_);
            s0 = wmma_bf16(qa0, b0, s0);
            s0 = wmma_bf16(qa1, b1, s0);
            v16bf b2 = load_b_fragT(Kb + (size_t)(j + 16) * HD_ + 0,  HD_);
            v16bf b3 = load_b_fragT(Kb + (size_t)(j + 16) * HD_ + 32, HD_);
            s1 = wmma_bf16(qa0, b2, s1);
            s1 = wmma_bf16(qa1, b3, s1);
        }
        if (j + 31 > q0) {                        // diagonal block: causal mask
#pragma unroll
            for (int r = 0; r < 8; ++r) {
                int query = q0 + mh + r;
                if (j + n      > query) s0[r] = -1e30f;
                if (j + 16 + n > query) s1[r] = -1e30f;
            }
        }
#pragma unroll
        for (int r = 0; r < 8; ++r) {
            float v = fmaxf(s0[r], s1[r]);
            v = fmaxf(v, __shfl_xor(v, 1, 32));
            v = fmaxf(v, __shfl_xor(v, 2, 32));
            v = fmaxf(v, __shfl_xor(v, 4, 32));
            v = fmaxf(v, __shfl_xor(v, 8, 32));
            float mn = fmaxf(mrow[r], v);
            float alpha = __expf(mrow[r] - mn);
            mrow[r] = mn;
            lrow[r] *= alpha;
            o[0][r] *= alpha; o[1][r] *= alpha; o[2][r] *= alpha; o[3][r] *= alpha;
            float p0 = __expf(s0[r] - mn);
            float p1 = __expf(s1[r] - mn);
            s0[r] = p0; s1[r] = p1;
            float ps = p0 + p1;
            ps += __shfl_xor(ps, 1, 32);
            ps += __shfl_xor(ps, 2, 32);
            ps += __shfl_xor(ps, 4, 32);
            ps += __shfl_xor(ps, 8, 32);
            lrow[r] += ps;
        }
        // C-layout -> A-layout bounce through LDS (same-wave DS ordering)
#pragma unroll
        for (int r = 0; r < 8; ++r) {
            myp[(mh + r) * 34 + n]      = f2bf(s0[r]);
            myp[(mh + r) * 34 + 16 + n] = f2bf(s1[r]);
        }
        v16bf pa = load_a_frag(myp, 34);
#pragma unroll
        for (int nt = 0; nt < 4; ++nt) {
            v16bf vfr = load_b_fragT(Vb + (size_t)(nt * 16) * L_ + j, L_);
            o[nt] = wmma_bf16(pa, vfr, o[nt]);
        }
    }

    size_t obase = ((size_t)b * L_ + q0) * D_ + (size_t)h * HD_;
#pragma unroll
    for (int r = 0; r < 8; ++r) {
        float inv = 1.0f / lrow[r];
        int m = mh + r;
#pragma unroll
        for (int nt = 0; nt < 4; ++nt)
            AO[obase + (size_t)m * D_ + nt * 16 + n] = f2bf(o[nt][r] * inv);
    }
}

// ---------- kernel 3: output projection, TDM-staged Wo ----------
// grid = (B*L/128, D/64), block = 256.

__global__ __launch_bounds__(256)
void mhsa_oproj(const __bf16* __restrict__ A, const __bf16* __restrict__ WoT,
                float* __restrict__ Out) {
    __shared__ __bf16 ldsB[2][64 * LDS_STRIDE];

    int wave = threadIdx.x >> 5;
    int lane = threadIdx.x & 31;
    int row0 = blockIdx.x * 128 + wave * 16;
    int col0 = blockIdx.y * 64;

    const __bf16* Arow  = A + (size_t)row0 * D_;
    const __bf16* Wbase = WoT + (size_t)col0 * D_;

    v8f acc[4] = {};
    if (wave == 0)
        tdm_load_2d(lds_off(&ldsB[0][0]), Wbase, D_, 1u << 20, D_, 32, 64);

    for (int ks = 0; ks < D_ / 32; ++ks) {
        int buf = ks & 1;
        int k0  = ks * 32;
        if (wave == 0) {
            if (ks + 1 < D_ / 32) {
                tdm_load_2d(lds_off(&ldsB[buf ^ 1][0]), Wbase + (ks + 1) * 32,
                            D_, 1u << 20, D_, 32, 64);
                __builtin_amdgcn_s_wait_tensorcnt(1);
            } else {
                __builtin_amdgcn_s_wait_tensorcnt(0);
            }
        }
        __syncthreads();

        if (k0 + 32 < D_) __builtin_prefetch(Arow + k0 + 32, 0, 1);
        v16bf a = load_a_frag(Arow + k0, D_);
#pragma unroll
        for (int nt = 0; nt < 4; ++nt)
            acc[nt] = wmma_bf16(a, load_b_fragT(&ldsB[buf][nt * 16 * LDS_STRIDE], LDS_STRIDE), acc[nt]);
        __syncthreads();
    }

    int n  = lane & 15;
    int mh = (lane & 16) ? 8 : 0;
#pragma unroll
    for (int nt = 0; nt < 4; ++nt)
#pragma unroll
        for (int r = 0; r < 8; ++r)
            Out[(size_t)(row0 + mh + r) * D_ + col0 + nt * 16 + n] = acc[nt][r];
}

// ---------- launcher ----------

extern "C" void kernel_launch(void* const* d_in, const int* in_sizes, int n_in,
                              void* d_out, int out_size, void* d_ws, size_t ws_size,
                              hipStream_t stream) {
    (void)in_sizes; (void)n_in; (void)out_size; (void)ws_size;
    const float* emb = (const float*)d_in[0];
    const float* Wq  = (const float*)d_in[1];
    const float* Wk  = (const float*)d_in[2];
    const float* Wv  = (const float*)d_in[3];
    const float* Wo  = (const float*)d_in[4];
    float* out = (float*)d_out;

    const size_t XB = (size_t)B_ * L_ * D_ * 2;        // 8 MB
    const size_t WT = (size_t)H_ * HD_ * D_ * 2;       // 2 MB
    const size_t WO = (size_t)D_ * D_ * 2;             // 2 MB
    const size_t QB = (size_t)B_ * H_ * L_ * HD_ * 2;  // 8 MB

    char* ws = (char*)d_ws;
    __bf16* Xbf = (__bf16*)(ws);
    __bf16* Wqt = (__bf16*)(ws + XB);
    __bf16* Wkt = (__bf16*)(ws + XB + WT);
    __bf16* Wvt = (__bf16*)(ws + XB + 2 * WT);
    __bf16* WoT = (__bf16*)(ws + XB + 3 * WT);
    __bf16* Qw  = (__bf16*)(ws + XB + 3 * WT + WO);
    __bf16* Kw  = (__bf16*)(ws + XB + 3 * WT + WO + QB);
    __bf16* Vtw = (__bf16*)(ws + XB + 3 * WT + WO + 2 * QB);
    __bf16* AOw = (__bf16*)(ws + XB + 3 * WT + WO + 3 * QB);

    mhsa_convert<<<1024, 256, 0, stream>>>(emb, Wq, Wk, Wv, Wo, Xbf, Wqt, Wkt, Wvt, WoT);
    mhsa_qkv<<<dim3(B_ * H_, L_ / 128), 256, 0, stream>>>(Xbf, Wqt, Wkt, Wvt, Qw, Kw, Vtw);
    mhsa_attn<<<dim3(B_ * H_, L_ / 128), 256, 0, stream>>>(Qw, Kw, Vtw, AOw);
    mhsa_oproj<<<dim3((B_ * L_) / 128, D_ / 64), 256, 0, stream>>>(AOw, WoT, out);
}
